// BigramLanguageModel_59931973648450
// MI455X (gfx1250) — compile-verified, matrix operand order checked by reading, not run on hardware
//
#include <hip/hip_runtime.h>

#define T_SEQ 1024
#define EMB   768
#define NH    6
#define HDIM  128
#define VOCAB 32000
#define NLAYER 6
#define MTOT  4096   // B*T

typedef __attribute__((ext_vector_type(16))) __bf16 v16bf;
typedef __attribute__((ext_vector_type(8)))  float  v8f;
typedef __attribute__((ext_vector_type(4)))  unsigned int v4u;
typedef __attribute__((ext_vector_type(4)))  int    v4i;
typedef __attribute__((ext_vector_type(8)))  int    v8i;

#if __has_builtin(__builtin_amdgcn_tensor_load_to_lds)
#define HAVE_TDM 1
#else
#define HAVE_TDM 0
#endif

__device__ inline __bf16 f2bf(float f) {
  unsigned u = __float_as_uint(f);
  unsigned r = u + 0x7FFFu + ((u >> 16) & 1u);   // round-to-nearest-even
  unsigned short h = (unsigned short)(r >> 16);
  return __builtin_bit_cast(__bf16, h);
}

__device__ inline v8f zero8() {
  v8f z = {0.f, 0.f, 0.f, 0.f, 0.f, 0.f, 0.f, 0.f};
  return z;
}

union FragU { uint4 q[2]; v16bf v; };

// 16x32 bf16 fragment in CDNA5 WMMA A/B layout from row-major storage:
// lane r (r=lane&15), hf=lane>>4: elems base[r*stride + hf*8 + 0..7] and
// base[r*stride + 16 + hf*8 + 0..7].
__device__ inline v16bf load_frag(const __bf16* base, int stride, int lane) {
  const int hf = lane >> 4, r = lane & 15;
  const __bf16* p = base + (size_t)r * stride + hf * 8;
  FragU u;
  u.q[0] = *(const uint4*)(p);
  u.q[1] = *(const uint4*)(p + 16);
  return u.v;
}

__device__ inline v8f wmma_bf16(v16bf a, v16bf b, v8f c) {
  return __builtin_amdgcn_wmma_f32_16x16x32_bf16(
      false, a, false, b, (short)0, c, false, false);
}

__device__ inline float redmax16(float v) {
  v = fmaxf(v, __shfl_xor(v, 1, 32));
  v = fmaxf(v, __shfl_xor(v, 2, 32));
  v = fmaxf(v, __shfl_xor(v, 4, 32));
  v = fmaxf(v, __shfl_xor(v, 8, 32));
  return v;
}
__device__ inline float redsum16(float v) {
  v += __shfl_xor(v, 1, 32);
  v += __shfl_xor(v, 2, 32);
  v += __shfl_xor(v, 4, 32);
  v += __shfl_xor(v, 8, 32);
  return v;
}

#if HAVE_TDM
// Issue a TDM load of a [tile_n=128 rows x tile_k=64 cols] bf16 tile from a
// row-major [N,K] tensor into LDS, inserting 16B of padding every 64B so the
// LDS row stride is 160B (80 elems) and every 16B fragment chunk stays aligned
// and bank-conflict free.  Descriptor bit layout per CDNA5 ISA.
__device__ inline void tdm_load_tile(unsigned lds_off, const __bf16* gaddr,
                                     int K, int N) {
  const unsigned long long ga = (unsigned long long)(size_t)gaddr;
  v4u g0;
  g0.x = 1u;                                             // count=1 (valid)
  g0.y = lds_off;                                        // lds_addr
  g0.z = (unsigned)(ga & 0xFFFFFFFFu);                   // global_addr[31:0]
  g0.w = (unsigned)((ga >> 32) & 0x1FFFFFFu) | (2u << 30);  // addr[56:32]|type=2
  v8i g1;
  // data_size=2B (code1), pad_enable, pad_interval=16DW (code3), pad_amount=4DW (code3)
  g1[0] = (int)((1u << 16) | (1u << 20) | (3u << 22) | (3u << 25));
  g1[1] = (int)(((unsigned)K & 0xFFFFu) << 16);          // tensor_dim0 lo16
  g1[2] = (int)(((unsigned)K >> 16) | (((unsigned)N & 0xFFFFu) << 16)); // dim0 hi | dim1 lo
  g1[3] = (int)(((unsigned)N >> 16) | (64u << 16));      // dim1 hi | tile_dim0=64
  g1[4] = (int)128u;                                     // tile_dim1=128, tile_dim2=0
  g1[5] = (int)(unsigned)K;                              // tensor_dim0_stride lo32
  g1[6] = 0;                                             // stride hi | dim1_stride lo
  g1[7] = 0;
  const v4i z4 = {0, 0, 0, 0};
#if defined(__clang_major__) && (__clang_major__ >= 23)
  const v8i z8 = {0, 0, 0, 0, 0, 0, 0, 0};
  __builtin_amdgcn_tensor_load_to_lds(g0, g1, z4, z4, z8, 0);
#else
  __builtin_amdgcn_tensor_load_to_lds(g0, g1, z4, z4, 0);
#endif
}
#endif  // HAVE_TDM

// ---------------------------------------------------------------- embedding
__global__ __launch_bounds__(256) void embed_kernel(
    const int* __restrict__ idx, const float* __restrict__ tok,
    const float* __restrict__ pos, float* __restrict__ x) {
  const int row = blockIdx.x, tid = threadIdx.x;
  const int t = row & (T_SEQ - 1);
  const int token = idx[row];
  const float* te = tok + (size_t)token * EMB;
  const float* pe = pos + (size_t)t * EMB;
  float* xp = x + (size_t)row * EMB;
  xp[tid]       = te[tid]       + pe[tid];
  xp[tid + 256] = te[tid + 256] + pe[tid + 256];
  xp[tid + 512] = te[tid + 512] + pe[tid + 512];
}

// ---------------------------------------------------------------- layernorm
__global__ __launch_bounds__(256) void layernorm_bf16(
    const float* __restrict__ x, const float* __restrict__ s,
    const float* __restrict__ b, __bf16* __restrict__ out) {
  __shared__ float red[256];
  const int row = blockIdx.x, tid = threadIdx.x;
  const float* xp = x + (size_t)row * EMB;
  float v0 = xp[tid], v1 = xp[tid + 256], v2 = xp[tid + 512];
  red[tid] = v0 + v1 + v2;
  __syncthreads();
  for (int o = 128; o > 0; o >>= 1) {
    if (tid < o) red[tid] += red[tid + o];
    __syncthreads();
  }
  const float mu = red[0] * (1.f / EMB);
  __syncthreads();
  const float d0 = v0 - mu, d1 = v1 - mu, d2 = v2 - mu;
  red[tid] = d0 * d0 + d1 * d1 + d2 * d2;
  __syncthreads();
  for (int o = 128; o > 0; o >>= 1) {
    if (tid < o) red[tid] += red[tid + o];
    __syncthreads();
  }
  const float rs = rsqrtf(red[0] * (1.f / EMB) + 1e-5f);
  __bf16* op = out + (size_t)row * EMB;
  op[tid]       = f2bf(d0 * rs * s[tid]       + b[tid]);
  op[tid + 256] = f2bf(d1 * rs * s[tid + 256] + b[tid + 256]);
  op[tid + 512] = f2bf(d2 * rs * s[tid + 512] + b[tid + 512]);
}

// ---------------------------------------------------------------- weight prep
// Transpose-convert W[K,N] f32 -> Wt[N,K] bf16 (tiled through LDS, coalesced
// on both sides). Done once per weight per forward (each weight is consumed
// by exactly one GEMM), so total traffic is unchanged while the GEMM hot loop
// reads half the bytes and the TDM can stream tiles whose rows are contiguous.
__global__ __launch_bounds__(256) void transpose_convert(
    const float* __restrict__ in, __bf16* __restrict__ out, int K, int N) {
  __shared__ float t[32][33];
  const int tid = threadIdx.x;
  const int nt = blockIdx.x * 32, kt = blockIdx.y * 32;
#pragma unroll
  for (int i = 0; i < 4; ++i) {
    const int lin = i * 256 + tid;
    const int r = lin >> 5, c = lin & 31;  // r = k-local, c = n-local
    t[r][c] = in[(size_t)(kt + r) * N + nt + c];
  }
  __syncthreads();
#pragma unroll
  for (int i = 0; i < 4; ++i) {
    const int lin = i * 256 + tid;
    const int r = lin >> 5, c = lin & 31;  // r = n-local, c = k-local
    out[(size_t)(nt + r) * K + kt + c] = f2bf(t[c][r]);
  }
}

// ---------------------------------------------------------------- GEMM
// C[M,N] = A[M,K](bf16,row-major) @ Wt[N,K](bf16,row-major==W^T) + bias.
// Block = 8 waves (2M x 4N), block tile 64x128, K-step 64.
// Wt tiles streamed into double-buffered LDS by the Tensor Data Mover
// (padded stride 80 elems = 160B rows), overlapped with WMMA compute.
enum { EPI_F32 = 0, EPI_F32_RES = 1, EPI_BF16_RELU = 2, EPI_HEADS = 3, EPI_HEADS_T = 4 };

#define LDS_WROW 80            // padded row stride in bf16 elems (160B)
#define LDS_WBUF (128 * LDS_WROW)

template <int EPI>
__global__ __launch_bounds__(256) void gemm_bf16(
    const __bf16* __restrict__ A, const __bf16* __restrict__ Wt,
    const float* __restrict__ bias, void* __restrict__ outp,
    const float* __restrict__ res, int N, int K) {
  __shared__ __bf16 ldsW[2][LDS_WBUF];
  const int tid = threadIdx.x;
  const int lane = tid & 31, wave = tid >> 5;
  const int wm = wave >> 2, wn = wave & 3;          // 2 (M) x 4 (N) waves
  const int m0 = blockIdx.y * 64 + wm * 32;
  const int n0b = blockIdx.x * 128;
  const int n0 = n0b + wn * 32;

  v8f acc[2][2];
  acc[0][0] = zero8(); acc[0][1] = zero8();
  acc[1][0] = zero8(); acc[1][1] = zero8();

#if HAVE_TDM
  if (wave == 0)
    tdm_load_tile((unsigned)(size_t)&ldsW[0][0],
                  Wt + (size_t)n0b * K, K, N);
#endif

  int cur = 0;
  for (int k0 = 0; k0 < K; k0 += 64, cur ^= 1) {
#if HAVE_TDM
    const bool more = (k0 + 64) < K;
    if (wave == 0) {
      if (more) {
        tdm_load_tile((unsigned)(size_t)&ldsW[cur ^ 1][0],
                      Wt + (size_t)n0b * K + k0 + 64, K, N);
        __builtin_amdgcn_s_wait_tensorcnt(1);
      } else {
        __builtin_amdgcn_s_wait_tensorcnt(0);
      }
    }
#else
    cur = 0;
    // cooperative fallback: copy Wt[n0b..+128, k0..+64] with padded stride
#pragma unroll
    for (int i = 0; i < 32; ++i) {
      const int lin = i * 256 + tid;
      const int n = lin >> 6, kk = lin & 63;
      ldsW[0][n * LDS_WROW + (kk >= 32 ? 40 + (kk - 32) : kk)] =
          Wt[(size_t)(n0b + n) * K + k0 + kk];
    }
#endif
    __syncthreads();
#pragma unroll
    for (int c = 0; c < 2; ++c) {
      const v16bf a0 = load_frag(A + (size_t)m0 * K + k0 + c * 32, K, lane);
      const v16bf a1 = load_frag(A + (size_t)(m0 + 16) * K + k0 + c * 32, K, lane);
      const v16bf b0 = load_frag(&ldsW[cur][(wn * 32) * LDS_WROW + c * 40], LDS_WROW, lane);
      const v16bf b1 = load_frag(&ldsW[cur][(wn * 32 + 16) * LDS_WROW + c * 40], LDS_WROW, lane);
      acc[0][0] = wmma_bf16(a0, b0, acc[0][0]);
      acc[0][1] = wmma_bf16(a0, b1, acc[0][1]);
      acc[1][0] = wmma_bf16(a1, b0, acc[1][0]);
      acc[1][1] = wmma_bf16(a1, b1, acc[1][1]);
    }
    __syncthreads();
  }

  const int nl = lane & 15, hf = lane >> 4;
#pragma unroll
  for (int mi = 0; mi < 2; ++mi)
#pragma unroll
    for (int ni = 0; ni < 2; ++ni)
#pragma unroll
      for (int v = 0; v < 8; ++v) {
        const int row = m0 + mi * 16 + hf * 8 + v;
        const int col = n0 + ni * 16 + nl;
        const float val = acc[mi][ni][v] + bias[col];
        if constexpr (EPI == EPI_F32) {
          ((float*)outp)[(size_t)row * N + col] = val;
        } else if constexpr (EPI == EPI_F32_RES) {
          ((float*)outp)[(size_t)row * N + col] =
              res[(size_t)row * N + col] + val;
        } else if constexpr (EPI == EPI_BF16_RELU) {
          ((__bf16*)outp)[(size_t)row * N + col] = f2bf(fmaxf(val, 0.f));
        } else {
          const int bb = row >> 10, tt = row & (T_SEQ - 1);
          const int hh = col >> 7, dd = col & (HDIM - 1);
          if constexpr (EPI == EPI_HEADS)
            ((__bf16*)outp)[(((size_t)bb * NH + hh) * T_SEQ + tt) * HDIM + dd] = f2bf(val);
          else  // EPI_HEADS_T: V stored [B,H,HD,T] so P@V B-frags are contiguous
            ((__bf16*)outp)[(((size_t)bb * NH + hh) * HDIM + dd) * T_SEQ + tt] = f2bf(val);
        }
      }
}

// ---------------------------------------------------------------- attention
// Flash-style causal attention. One wave per 16 query rows of one (b,h).
// 32 keys per iteration: 8 WMMAs for scores, online softmax, 8 WMMAs for O.
__global__ __launch_bounds__(256) void attn_kernel(
    const __bf16* __restrict__ q, const __bf16* __restrict__ k,
    const __bf16* __restrict__ vt, __bf16* __restrict__ o) {
  __shared__ __bf16 ldsP[8 * 16 * 40];  // wave-private P tile [16 rows][32 keys]
  const int lane = threadIdx.x & 31, wave = threadIdx.x >> 5;
  const int tile = blockIdx.x * 8 + wave;  // 0 .. B*H*(T/16)-1
  const int qb = tile & 63;
  const int bh = tile >> 6;
  const int q0 = qb * 16;
  const int nl = lane & 15, hf = lane >> 4;
  const __bf16* qp = q + ((size_t)bh * T_SEQ + q0) * HDIM;
  const __bf16* kp = k + (size_t)bh * T_SEQ * HDIM;
  const __bf16* vp = vt + (size_t)bh * HDIM * T_SEQ;
  __bf16* pb = ldsP + wave * 16 * 40;

  v16bf aq[4];
#pragma unroll
  for (int c = 0; c < 4; ++c) aq[c] = load_frag(qp + c * 32, HDIM, lane);

  v8f accO[8];
#pragma unroll
  for (int nt = 0; nt < 8; ++nt) accO[nt] = zero8();
  float mrow[8], lrow[8];
#pragma unroll
  for (int v = 0; v < 8; ++v) { mrow[v] = -1e30f; lrow[v] = 0.f; }

  const float scale = 0.08838834764831845f;  // 1/sqrt(128)
  const int kblocks = (q0 + 15) / 32 + 1;

  for (int kb = 0; kb < kblocks; ++kb) {
    const int kbase = kb * 32;
    v8f s0 = zero8(), s1 = zero8();
#pragma unroll
    for (int c = 0; c < 4; ++c) {
      const v16bf bk0 = load_frag(kp + (size_t)kbase * HDIM + c * 32, HDIM, lane);
      const v16bf bk1 = load_frag(kp + (size_t)(kbase + 16) * HDIM + c * 32, HDIM, lane);
      s0 = wmma_bf16(aq[c], bk0, s0);
      s1 = wmma_bf16(aq[c], bk1, s1);
    }
    float p0[8], p1[8], corr[8];
#pragma unroll
    for (int v = 0; v < 8; ++v) {
      const int qidx = q0 + hf * 8 + v;
      float sa = s0[v] * scale;
      if (kbase + nl > qidx) sa = -1e30f;          // causal mask
      float sb = s1[v] * scale;
      if (kbase + 16 + nl > qidx) sb = -1e30f;
      const float rm = redmax16(fmaxf(sa, sb));
      const float mnew = fmaxf(mrow[v], rm);
      corr[v] = __expf(mrow[v] - mnew);
      p0[v] = __expf(sa - mnew);
      p1[v] = __expf(sb - mnew);
      const float rs = redsum16(p0[v] + p1[v]);
      lrow[v] = lrow[v] * corr[v] + rs;
      mrow[v] = mnew;
    }
#pragma unroll
    for (int nt = 0; nt < 8; ++nt)
#pragma unroll
      for (int v = 0; v < 8; ++v) accO[nt][v] *= corr[v];

    // transpose P from C-layout to A-layout through wave-private LDS
#pragma unroll
    for (int v = 0; v < 8; ++v) {
      pb[(hf * 8 + v) * 40 + nl]      = f2bf(p0[v]);
      pb[(hf * 8 + v) * 40 + 16 + nl] = f2bf(p1[v]);
    }
    asm volatile("s_wait_dscnt 0" ::: "memory");
    const v16bf ap = load_frag(pb, 40, lane);
#pragma unroll
    for (int nt = 0; nt < 8; ++nt) {
      const v16bf bv = load_frag(vp + (size_t)(nt * 16) * T_SEQ + kbase, T_SEQ, lane);
      accO[nt] = wmma_bf16(ap, bv, accO[nt]);
    }
  }

  const int bb = bh / NH, hh = bh % NH;
#pragma unroll
  for (int nt = 0; nt < 8; ++nt)
#pragma unroll
    for (int v = 0; v < 8; ++v) {
      const int tt = q0 + hf * 8 + v;
      const int dd = nt * 16 + nl;
      const float ov = accO[nt][v] / lrow[v];
      o[((size_t)bb * T_SEQ + tt) * EMB + hh * HDIM + dd] = f2bf(ov);
    }
}

// ---------------------------------------------------------------- loss
__global__ __launch_bounds__(256) void rowloss_kernel(
    const float* __restrict__ logits, const int* __restrict__ targets,
    float* __restrict__ rl) {
  __shared__ float red[256];
  const int row = blockIdx.x, tid = threadIdx.x;
  const float* lp = logits + (size_t)row * VOCAB;
  float mx = -1e30f;
  for (int i = tid; i < VOCAB; i += 256) mx = fmaxf(mx, lp[i]);
  red[tid] = mx;
  __syncthreads();
  for (int o = 128; o > 0; o >>= 1) {
    if (tid < o) red[tid] = fmaxf(red[tid], red[tid + o]);
    __syncthreads();
  }
  mx = red[0];
  __syncthreads();
  float se = 0.f;
  for (int i = tid; i < VOCAB; i += 256) se += __expf(lp[i] - mx);
  red[tid] = se;
  __syncthreads();
  for (int o = 128; o > 0; o >>= 1) {
    if (tid < o) red[tid] += red[tid + o];
    __syncthreads();
  }
  if (tid == 0) {
    const float lse = mx + logf(red[0]);
    rl[row] = lse - lp[targets[row]];
  }
}

__global__ __launch_bounds__(256) void loss_reduce_kernel(
    const float* __restrict__ rl, float* __restrict__ out) {
  __shared__ float red[256];
  const int tid = threadIdx.x;
  float s = 0.f;
  for (int i = tid; i < MTOT; i += 256) s += rl[i];
  red[tid] = s;
  __syncthreads();
  for (int o = 128; o > 0; o >>= 1) {
    if (tid < o) red[tid] += red[tid + o];
    __syncthreads();
  }
  if (tid == 0) out[0] = red[0] * (1.f / MTOT);
}

// ---------------------------------------------------------------- host
extern "C" void kernel_launch(void* const* d_in, const int* in_sizes, int n_in,
                              void* d_out, int out_size, void* d_ws, size_t ws_size,
                              hipStream_t stream) {
  const int*   idx     = (const int*)d_in[0];
  const int*   targets = (const int*)d_in[1];
  const float* tok_emb = (const float*)d_in[2];
  const float* pos_emb = (const float*)d_in[3];
  const float* Wq = (const float*)d_in[4];
  const float* bq = (const float*)d_in[5];
  const float* Wk = (const float*)d_in[6];
  const float* bk = (const float*)d_in[7];
  const float* Wv = (const float*)d_in[8];
  const float* bv = (const float*)d_in[9];
  const float* Wo = (const float*)d_in[10];
  const float* bo = (const float*)d_in[11];
  const float* w1 = (const float*)d_in[12];
  const float* b1 = (const float*)d_in[13];
  const float* w2 = (const float*)d_in[14];
  const float* b2 = (const float*)d_in[15];
  const float* ln1_s = (const float*)d_in[16];
  const float* ln1_b = (const float*)d_in[17];
  const float* ln2_s = (const float*)d_in[18];
  const float* ln2_b = (const float*)d_in[19];
  const float* lnf_s = (const float*)d_in[20];
  const float* lnf_b = (const float*)d_in[21];
  const float* Wlm = (const float*)d_in[22];
  const float* blm = (const float*)d_in[23];
  (void)in_sizes; (void)n_in; (void)out_size; (void)ws_size;

  char* ws = (char*)d_ws;
  size_t off = 0;
  auto alloc = [&](size_t bytes) -> void* {
    void* p = ws + off;
    off = (off + bytes + 255) & ~(size_t)255;
    return p;
  };
  float*  x      = (float*)alloc((size_t)MTOT * EMB * sizeof(float));
  __bf16* hb     = (__bf16*)alloc((size_t)MTOT * EMB * 2);
  __bf16* qbuf   = (__bf16*)alloc((size_t)MTOT * EMB * 2);
  __bf16* kbuf   = (__bf16*)alloc((size_t)MTOT * EMB * 2);
  __bf16* vtbuf  = (__bf16*)alloc((size_t)MTOT * EMB * 2);
  __bf16* obuf   = (__bf16*)alloc((size_t)MTOT * EMB * 2);
  __bf16* midbuf = (__bf16*)alloc((size_t)MTOT * 4 * EMB * 2);
  __bf16* wtbuf  = (__bf16*)alloc((size_t)VOCAB * EMB * 2);  // max weight (Wlm)
  float*  rloss  = (float*)alloc((size_t)MTOT * sizeof(float));

  const dim3 blk(256);
  const dim3 gE(EMB / 128, MTOT / 64);        // N=768 GEMMs
  const dim3 gM(4 * EMB / 128, MTOT / 64);    // N=3072 GEMM
  const dim3 gV(VOCAB / 128, MTOT / 64);      // N=32000 GEMM
  const dim3 cE(EMB / 32, EMB / 32);          // transpose K=768,N=768
  const dim3 cM1(4 * EMB / 32, EMB / 32);     // K=768,N=3072
  const dim3 cM2(EMB / 32, 4 * EMB / 32);     // K=3072,N=768
  const dim3 cV(VOCAB / 32, EMB / 32);        // K=768,N=32000

  embed_kernel<<<MTOT, blk, 0, stream>>>(idx, tok_emb, pos_emb, x);

  for (int l = 0; l < NLAYER; ++l) {
    const size_t wE = (size_t)l * EMB * EMB;
    const size_t wF = (size_t)l * EMB * 4 * EMB;
    layernorm_bf16<<<MTOT, blk, 0, stream>>>(x, ln1_s + l * EMB, ln1_b + l * EMB, hb);

    transpose_convert<<<cE, blk, 0, stream>>>(Wq + wE, wtbuf, EMB, EMB);
    gemm_bf16<EPI_HEADS><<<gE, blk, 0, stream>>>(hb, wtbuf, bq + l * EMB, qbuf, nullptr, EMB, EMB);
    transpose_convert<<<cE, blk, 0, stream>>>(Wk + wE, wtbuf, EMB, EMB);
    gemm_bf16<EPI_HEADS><<<gE, blk, 0, stream>>>(hb, wtbuf, bk + l * EMB, kbuf, nullptr, EMB, EMB);
    transpose_convert<<<cE, blk, 0, stream>>>(Wv + wE, wtbuf, EMB, EMB);
    gemm_bf16<EPI_HEADS_T><<<gE, blk, 0, stream>>>(hb, wtbuf, bv + l * EMB, vtbuf, nullptr, EMB, EMB);

    attn_kernel<<<(4 * NH * (T_SEQ / 16)) / 8, blk, 0, stream>>>(qbuf, kbuf, vtbuf, obuf);

    transpose_convert<<<cE, blk, 0, stream>>>(Wo + wE, wtbuf, EMB, EMB);
    gemm_bf16<EPI_F32_RES><<<gE, blk, 0, stream>>>(obuf, wtbuf, bo + l * EMB, x, x, EMB, EMB);

    layernorm_bf16<<<MTOT, blk, 0, stream>>>(x, ln2_s + l * EMB, ln2_b + l * EMB, hb);

    transpose_convert<<<cM1, blk, 0, stream>>>(w1 + wF, wtbuf, EMB, 4 * EMB);
    gemm_bf16<EPI_BF16_RELU><<<gM, blk, 0, stream>>>(hb, wtbuf, b1 + l * 4 * EMB, midbuf, nullptr, 4 * EMB, EMB);
    transpose_convert<<<cM2, blk, 0, stream>>>(w2 + wF, wtbuf, 4 * EMB, EMB);
    gemm_bf16<EPI_F32_RES><<<gE, blk, 0, stream>>>(midbuf, wtbuf, b2 + l * EMB, x, x, EMB, 4 * EMB);
  }

  layernorm_bf16<<<MTOT, blk, 0, stream>>>(x, lnf_s, lnf_b, hb);
  float* logits = (float*)d_out;
  transpose_convert<<<cV, blk, 0, stream>>>(Wlm, wtbuf, EMB, VOCAB);
  gemm_bf16<EPI_F32><<<gV, blk, 0, stream>>>(hb, wtbuf, blm, logits, nullptr, VOCAB, EMB);
  rowloss_kernel<<<MTOT, blk, 0, stream>>>(logits, targets, rloss);
  loss_reduce_kernel<<<1, blk, 0, stream>>>(rloss, logits + (size_t)MTOT * VOCAB);
}
